// lstm_source_65085934404071
// MI455X (gfx1250) — matrix operand order repaired and checked by reading, not compiled, not run on hardware
//
#include <hip/hip_runtime.h>
#include <hip/hip_bf16.h>

#define HDIM 1024
#define BDIM 32
#define TDIM 256
#define NLAYERS 4
#define GDIM (4 * HDIM)   // 4096 gate columns

typedef __attribute__((ext_vector_type(16))) __bf16 v16bf;
typedef __attribute__((ext_vector_type(8)))  __bf16 v8bf;
typedef __attribute__((ext_vector_type(8)))  float  v8f;

// ---------------------------------------------------------------------------
// A-operand loader for V_WMMA_F32_16X16X32_BF16.
// Layout (matches packed B): lanes 0-15 -> M=lane, K = {0..15};
// lanes 16-31 -> M=lane-16, K = {16..31}.  So each lane needs one
// contiguous 16-element run starting at k0+16g (loaded as two 16B halves).
// ---------------------------------------------------------------------------
__device__ __forceinline__ v16bf load_a_tile(const __bf16* rowk0, int g) {
  v8bf lo = *(const v8bf*)(rowk0 + (g << 4));
  v8bf hi = *(const v8bf*)(rowk0 + 8 + (g << 4));
  return __builtin_shufflevector(lo, hi, 0, 1, 2, 3, 4, 5, 6, 7,
                                 8, 9, 10, 11, 12, 13, 14, 15);
}

// ---------------------------------------------------------------------------
// Pack fp32 [K,N] row-major weights into bf16 WMMA-B tile layout:
// tile (kt,nt) covers K=[kt*32,kt*32+32), N=[nt*16,nt*16+16).
// lane l of tile holds 16 contiguous bf16: element j = W[kt*32+16*(l/16)+j][nt*16+l%16].
// One thread per (tile, lane): 16 strided reads, 32B contiguous write.
// ---------------------------------------------------------------------------
__global__ void pack_b_kernel(const float* __restrict__ src,
                              __bf16* __restrict__ dst, int K, int N) {
  int KT = K >> 5, NT = N >> 4;
  int idx = blockIdx.x * blockDim.x + threadIdx.x;
  if (idx >= KT * NT * 32) return;
  int lane = idx & 31;
  int tile = idx >> 5;
  int nt = tile % NT, kt = tile / NT;
  int g = lane >> 4;
  int n = (nt << 4) + (lane & 15);
  int kbase = (kt << 5) + (g << 4);
  __bf16* out = dst + (size_t)tile * 512 + lane * 16;
#pragma unroll
  for (int j = 0; j < 16; ++j)
    out[j] = (__bf16)src[(size_t)(kbase + j) * N + n];
}

// ---------------------------------------------------------------------------
// Convert x [B,T,H] f32 -> seq [T,B,H] bf16; init h/c state from h0/c0.
// ---------------------------------------------------------------------------
__global__ void init_state_kernel(const float* __restrict__ x,
                                  const float* __restrict__ h0,
                                  const float* __restrict__ c0,
                                  __bf16* __restrict__ seq,
                                  __bf16* __restrict__ h0bf,
                                  float* __restrict__ hf,
                                  float* __restrict__ cf) {
  size_t i = (size_t)blockIdx.x * blockDim.x + threadIdx.x;
  size_t total = (size_t)TDIM * BDIM * HDIM;
  if (i < total) {
    int hh = (int)(i % HDIM);
    size_t tb = i / HDIM;
    int bb = (int)(tb % BDIM);
    int tt = (int)(tb / BDIM);
    seq[i] = (__bf16)x[((size_t)bb * TDIM + tt) * HDIM + hh];
  }
  if (i < (size_t)BDIM * HDIM) {
    h0bf[i] = (__bf16)h0[i];
    hf[i] = h0[i];
    cf[i] = c0[i];
  }
}

// ---------------------------------------------------------------------------
// Generic bf16 WMMA GEMM: C[M,N] = A[M,K] @ Bpk + bias.
// Each wave: 16 rows x 64 cols (4 accumulator tiles sharing the A tile).
// MODE 0: store bf16 row-major [M,N] (xw projections, next-layer input).
// MODE 1: store f32, rows m = t*B+b remapped to out[(b*T+t)*N + n] (head).
// ---------------------------------------------------------------------------
template <int MODE>
__global__ void wmma_gemm_kernel(const __bf16* __restrict__ A,
                                 const __bf16* __restrict__ Bpk,
                                 const float* __restrict__ bias,
                                 void* __restrict__ Out,
                                 int Mdim, int Kdim, int Ndim) {
  int gwave = (blockIdx.x * blockDim.x + threadIdx.x) >> 5;
  int lane = threadIdx.x & 31;
  int nGroups = Ndim >> 6;
  int mt = gwave / nGroups;
  int ng = gwave % nGroups;
  if (mt >= (Mdim >> 4)) return;
  int m0 = mt << 4, n0 = ng << 6;
  int g = lane >> 4, ln = lane & 15;
  int NT = Ndim >> 4;
  v8f acc[4] = {};
  const __bf16* arow = A + (size_t)(m0 + ln) * Kdim;
  int KT = Kdim >> 5;
  for (int kt = 0; kt < KT; ++kt) {
    v16bf a = load_a_tile(arow + (kt << 5), g);
    const __bf16* bbase = Bpk + ((size_t)kt * NT + (n0 >> 4)) * 512 + lane * 16;
    __builtin_prefetch(bbase + (size_t)NT * 512, 0, 1);  // next K-tile strip
#pragma unroll
    for (int q = 0; q < 4; ++q) {
      v16bf bv = *(const v16bf*)(bbase + q * 512);
      acc[q] = __builtin_amdgcn_wmma_f32_16x16x32_bf16(
          false, a, false, bv, (short)0, acc[q], false, false);
    }
  }
#pragma unroll
  for (int q = 0; q < 4; ++q) {
    int n = n0 + (q << 4) + ln;
    float bv = bias[n];
#pragma unroll
    for (int r = 0; r < 8; ++r) {
      int m = m0 + (g << 3) + r;
      float v = acc[q][r] + bv;
      if (MODE == 0) {
        ((__bf16*)Out)[(size_t)m * Ndim + n] = (__bf16)v;
      } else {
        int tt = m / BDIM, bb = m % BDIM;
        ((float*)Out)[((size_t)bb * TDIM + tt) * Ndim + n] = v;
      }
    }
  }
}

// ---------------------------------------------------------------------------
// One recurrent step:  z = xw_t + h @ Wr ; gates; update c; emit h.
// Grid: 128 blocks (2 M-tiles x 64 N-tiles), 4 waves/block split K (256 each).
// Each wave accumulates the same 16x16 slice of all four gates (i,f,g,o share
// the A tile), partials reduced through LDS, wave 0 runs the gate math.
// seq[t] doubles as the h ping-pong buffer.
// ---------------------------------------------------------------------------
__global__ void lstm_step_kernel(const __bf16* __restrict__ xw_t,  // [B,4H]
                                 const __bf16* __restrict__ Wr_pk,
                                 float* __restrict__ cbuf,         // [B,H] in/out
                                 const __bf16* __restrict__ hin,   // [B,H]
                                 float* __restrict__ hf,           // [B,H] f32 h
                                 __bf16* __restrict__ seq_t) {     // [B,H] bf16 h
  __shared__ float red[4][4][32][8];  // [wave][gate][lane][r] = 16 KB
  int wave = threadIdx.x >> 5;
  int lane = threadIdx.x & 31;
  int mt = blockIdx.x >> 6;   // 0..1
  int nt = blockIdx.x & 63;   // 0..63
  int m0 = mt << 4, n0 = nt << 4;
  int g = lane >> 4, ln = lane & 15;
  v8f acc[4] = {};
  const int NT = GDIM >> 4;   // 256 N-tiles in packed Wr
  int kt0 = wave << 3;        // 8 K-tiles per wave
  const __bf16* arow = hin + (size_t)(m0 + ln) * HDIM;
#pragma unroll
  for (int kk = 0; kk < 8; ++kk) {
    int kt = kt0 + kk;
    v16bf a = load_a_tile(arow + (kt << 5), g);
#pragma unroll
    for (int q = 0; q < 4; ++q) {
      int ntile = q * (HDIM >> 4) + nt;  // gate q at column offset q*H
      const v16bf* bp =
          (const v16bf*)(Wr_pk + ((size_t)kt * NT + ntile) * 512 + lane * 16);
      acc[q] = __builtin_amdgcn_wmma_f32_16x16x32_bf16(
          false, a, false, *bp, (short)0, acc[q], false, false);
    }
  }
#pragma unroll
  for (int q = 0; q < 4; ++q)
#pragma unroll
    for (int r = 0; r < 8; ++r) red[wave][q][lane][r] = acc[q][r];
  __syncthreads();
  if (wave == 0) {
#pragma unroll
    for (int r = 0; r < 8; ++r) {
      int m = m0 + (g << 3) + r;
      int n = n0 + ln;
      const __bf16* xwrow = xw_t + (size_t)m * GDIM + n;
      float zi = red[0][0][lane][r] + red[1][0][lane][r] +
                 red[2][0][lane][r] + red[3][0][lane][r] + (float)xwrow[0 * HDIM];
      float zf = red[0][1][lane][r] + red[1][1][lane][r] +
                 red[2][1][lane][r] + red[3][1][lane][r] + (float)xwrow[1 * HDIM];
      float zg = red[0][2][lane][r] + red[1][2][lane][r] +
                 red[2][2][lane][r] + red[3][2][lane][r] + (float)xwrow[2 * HDIM];
      float zo = red[0][3][lane][r] + red[1][3][lane][r] +
                 red[2][3][lane][r] + red[3][3][lane][r] + (float)xwrow[3 * HDIM];
      float si = 1.f / (1.f + __expf(-zi));
      float sf = 1.f / (1.f + __expf(-zf));
      float so = 1.f / (1.f + __expf(-zo));
      float gg = fmaxf(zg, 0.f);
      size_t idx = (size_t)m * HDIM + n;
      float cn = sf * cbuf[idx] + si * gg;
      float hn = so * fmaxf(cn, 0.f);
      cbuf[idx] = cn;
      hf[idx] = hn;
      seq_t[idx] = (__bf16)hn;
    }
  }
}

__global__ void tail_copy_kernel(const float* __restrict__ hf,
                                 const float* __restrict__ cf,
                                 float* __restrict__ out, int n) {
  int i = blockIdx.x * blockDim.x + threadIdx.x;
  if (i < n) {
    out[i] = hf[i];
    out[n + i] = cf[i];
  }
}

extern "C" void kernel_launch(void* const* d_in, const int* in_sizes, int n_in,
                              void* d_out, int out_size, void* d_ws, size_t ws_size,
                              hipStream_t stream) {
  const float* x  = (const float*)d_in[0];
  const float* h0 = (const float*)d_in[1];
  const float* c0 = (const float*)d_in[2];
  const float* Wk = (const float*)d_in[3];
  const float* Wr = (const float*)d_in[4];
  const float* bg = (const float*)d_in[5];
  const float* Wd = (const float*)d_in[6];
  const float* bd = (const float*)d_in[7];

  char* ws = (char*)d_ws;
  size_t off = 0;
  __bf16* wk_pk = (__bf16*)(ws + off); off += (size_t)NLAYERS * HDIM * GDIM * 2;  // 32 MB
  __bf16* wr_pk = (__bf16*)(ws + off); off += (size_t)NLAYERS * HDIM * GDIM * 2;  // 32 MB
  __bf16* wd_pk = (__bf16*)(ws + off); off += (size_t)HDIM * HDIM * 2;            // 2 MB
  __bf16* seq   = (__bf16*)(ws + off); off += (size_t)TDIM * BDIM * HDIM * 2;     // 16 MB
  __bf16* xw    = (__bf16*)(ws + off); off += (size_t)TDIM * BDIM * GDIM * 2;     // 64 MB
  float*  cbuf  = (float*) (ws + off); off += (size_t)BDIM * HDIM * 4;
  float*  hf    = (float*) (ws + off); off += (size_t)BDIM * HDIM * 4;
  __bf16* h0bf  = (__bf16*)(ws + off); off += (size_t)BDIM * HDIM * 2;
  (void)ws_size; (void)in_sizes; (void)n_in; (void)out_size;

  // Pack weights to bf16 WMMA-B tile layout (L2-resident working set).
  {
    int thr = (HDIM >> 5) * (GDIM >> 4) * 32;  // 262144
    for (int l = 0; l < NLAYERS; ++l) {
      pack_b_kernel<<<thr / 256, 256, 0, stream>>>(
          Wk + (size_t)l * HDIM * GDIM, wk_pk + (size_t)l * HDIM * GDIM, HDIM, GDIM);
      pack_b_kernel<<<thr / 256, 256, 0, stream>>>(
          Wr + (size_t)l * HDIM * GDIM, wr_pk + (size_t)l * HDIM * GDIM, HDIM, GDIM);
    }
    int thrd = (HDIM >> 5) * (HDIM >> 4) * 32;  // 65536
    pack_b_kernel<<<thrd / 256, 256, 0, stream>>>(Wd, wd_pk, HDIM, HDIM);
  }
  {
    size_t total = (size_t)TDIM * BDIM * HDIM;
    init_state_kernel<<<(int)((total + 255) / 256), 256, 0, stream>>>(
        x, h0, c0, seq, h0bf, hf, cbuf);
  }

  const int MROWS = TDIM * BDIM;  // 8192
  for (int l = 0; l < NLAYERS; ++l) {
    // Input projection: xw = seq @ Wk[l] + b[l]  (bulk WMMA GEMM)
    int blocks = (MROWS >> 4) * (GDIM >> 6) / 4;  // 8192
    wmma_gemm_kernel<0><<<blocks, 128, 0, stream>>>(
        seq, wk_pk + (size_t)l * HDIM * GDIM, bg + (size_t)l * GDIM, xw,
        MROWS, HDIM, GDIM);
    // Sequential recurrent scan; seq[] is the h history (ping-pong by index).
    for (int t = 0; t < TDIM; ++t) {
      const __bf16* hin =
          (t == 0) ? (l == 0 ? h0bf : seq + (size_t)(TDIM - 1) * BDIM * HDIM)
                   : seq + (size_t)(t - 1) * BDIM * HDIM;
      lstm_step_kernel<<<128, 128, 0, stream>>>(
          xw + (size_t)t * BDIM * GDIM, wr_pk + (size_t)l * HDIM * GDIM,
          cbuf, hin, hf, seq + (size_t)t * BDIM * HDIM);
    }
  }

  // Dense head: out[b,t,:] = seq[t,b,:] @ Wd + bd  (f32, transposed store)
  {
    int blocks = (MROWS >> 4) * (HDIM >> 6) / 4;  // 2048
    wmma_gemm_kernel<1><<<blocks, 128, 0, stream>>>(
        seq, wd_pk, bd, d_out, MROWS, HDIM, HDIM);
  }
  // Final (hT, cT) of last layer appended after out.
  {
    float* tail = (float*)d_out + (size_t)BDIM * TDIM * HDIM;
    int n = BDIM * HDIM;
    tail_copy_kernel<<<(n + 255) / 256, 256, 0, stream>>>(hf, cbuf, tail, n);
  }
}
